// GemmaForCausalLM_39608188404313
// MI455X (gfx1250) — matrix operand order repaired and checked by reading, not compiled, not run on hardware
//
#include <hip/hip_runtime.h>
#include <cstdint>

// ---------------------------------------------------------------- types
typedef __bf16 bf16;
typedef __attribute__((ext_vector_type(16))) __bf16 v16bf;
typedef __attribute__((ext_vector_type(8)))  __bf16 v8bf;
typedef __attribute__((ext_vector_type(8)))  float  v8f;

#define WMMA_BF16(A, B, C) \
  __builtin_amdgcn_wmma_f32_16x16x32_bf16(false, (A), false, (B), (short)0, (C), false, false)

// CDNA5 async copy: 16B global -> LDS, tracked by ASYNCcnt (no VGPR data path)
__device__ __forceinline__ void async_copy16(const bf16* g, bf16* l) {
  uint32_t laddr = (uint32_t)(uintptr_t)l;     // LDS aperture: low 32 bits = LDS offset
  uint64_t gaddr = (uint64_t)(uintptr_t)g;
  asm volatile("global_load_async_to_lds_b128 %0, %1, off" :: "v"(laddr), "v"(gaddr) : "memory");
}
__device__ __forceinline__ void wait_async0() {
  asm volatile("s_wait_asynccnt 0x0" ::: "memory");
}

// ---------------------------------------------------------------- shapes
constexpr int LAYERS = 2;
constexpr int BB     = 2;
constexpr int SS     = 2048;
constexpr int HH     = 2048;
constexpr int NHEAD  = 8;
constexpr int HDIM   = 256;     // SCALE = 1/16 (folded into RoPE output)
constexpr int IDIM   = 16384;
constexpr int MTOK   = BB * SS;

// ================================================================ fp32 -> bf16 weight conversion
__global__ __launch_bounds__(256) void cvt_bf16_kernel(const float4* __restrict__ src,
                                                       bf16* __restrict__ dst, size_t n4) {
  size_t i = (size_t)blockIdx.x * blockDim.x + threadIdx.x;
  const size_t stride = (size_t)gridDim.x * blockDim.x;
  for (; i < n4; i += stride) {
    float4 v = src[i];
    bf16* d = dst + i * 4;
    d[0] = (bf16)v.x; d[1] = (bf16)v.y; d[2] = (bf16)v.z; d[3] = (bf16)v.w;
  }
}

// ================================================================ RMSNorm
template <typename OutT>
__global__ __launch_bounds__(256) void rmsnorm_kernel(const float* __restrict__ x,
                                                      const float* __restrict__ w,
                                                      OutT* __restrict__ out, int H) {
  const int row = blockIdx.x;
  const float* xr = x + (size_t)row * H;
  float ss = 0.f;
  for (int i = threadIdx.x; i < H; i += blockDim.x) { float v = xr[i]; ss += v * v; }
  __shared__ float red[8];
  for (int off = 16; off > 0; off >>= 1) ss += __shfl_down(ss, off, 32);
  const int wid = threadIdx.x >> 5, ln = threadIdx.x & 31;
  if (ln == 0) red[wid] = ss;
  __syncthreads();
  if (wid == 0) {
    float v = (ln < 8) ? red[ln] : 0.f;
    for (int off = 4; off > 0; off >>= 1) v += __shfl_down(v, off, 32);
    if (ln == 0) red[0] = rsqrtf(v / (float)H + 1e-6f);
  }
  __syncthreads();
  const float sc = red[0];
  OutT* orow = out + (size_t)row * H;
  for (int i = threadIdx.x; i < H; i += blockDim.x)
    orow[i] = (OutT)(xr[i] * sc * (1.0f + w[i]));
}

// ================================================================ GEMM
// out[M,N] (+)= A[M,K](bf16) x W[N,K]^T(bf16); async double-buffered LDS staging.
enum { GEMM_OUT_F32 = 0, GEMM_ACC_F32 = 1, GEMM_GATEUP = 2 };

template <int MODE>
__global__ __launch_bounds__(256) void gemm_kernel(const bf16* __restrict__ A,
                                                   const bf16* __restrict__ W0,
                                                   const bf16* __restrict__ W1,
                                                   float* __restrict__ outF,
                                                   bf16* __restrict__ outB,
                                                   int M, int N, int K) {
  constexpr int BM = 128, BN = 128, BK = 32, PK = 40;   // PK: padded LDS row
  __shared__ bf16 As[2][BM * PK];
  __shared__ bf16 Bs[2][BN * PK];
  __shared__ bf16 B2s[(MODE == GEMM_GATEUP) ? 2 * BN * PK : 1];

  const int m0 = blockIdx.y * BM, n0 = blockIdx.x * BN;
  const int tid = threadIdx.x, lane = tid & 31, w = tid >> 5;
  const int wm = (w & 3) * 32, wn = (w >> 2) * 64;      // wave sub-tile: 32 x 64
  const int lhalf = lane >> 4, l16 = lane & 15;
  const int grow = tid >> 1, ghalf = tid & 1;           // staging roles (32B per thread)

  v8f c0[2][4], c1[2][4];
#pragma unroll
  for (int i = 0; i < 2; i++)
#pragma unroll
    for (int j = 0; j < 4; j++) { c0[i][j] = v8f{}; if (MODE == GEMM_GATEUP) c1[i][j] = v8f{}; }

  auto issue = [&](int k0, int buf) {
    const bf16* asrc = A + (size_t)(m0 + grow) * K + k0 + ghalf * 16;
    bf16* adst = &As[buf][grow * PK + ghalf * 16];
    async_copy16(asrc, adst);
    async_copy16(asrc + 8, adst + 8);
    const bf16* bsrc = W0 + (size_t)(n0 + grow) * K + k0 + ghalf * 16;
    bf16* bdst = &Bs[buf][grow * PK + ghalf * 16];
    async_copy16(bsrc, bdst);
    async_copy16(bsrc + 8, bdst + 8);
    if (MODE == GEMM_GATEUP) {
      const bf16* csrc = W1 + (size_t)(n0 + grow) * K + k0 + ghalf * 16;
      bf16* cdst = &B2s[buf * BN * PK + grow * PK + ghalf * 16];
      async_copy16(csrc, cdst);
      async_copy16(csrc + 8, cdst + 8);
    }
  };

  const int nK = K / BK;
  issue(0, 0);
  for (int it = 0; it < nK; it++) {
    const int cur = it & 1;
    wait_async0();            // tile `it` fully landed (our own async ops, in-order)
    __syncthreads();          // ... for every wave in the workgroup
    if (it + 1 < nK) issue((it + 1) * BK, cur ^ 1);   // overlap next copy with WMMAs

    v16bf a[2];
#pragma unroll
    for (int mt = 0; mt < 2; mt++) {
      const bf16* rp = &As[cur][(wm + mt * 16 + l16) * PK];
      v8bf lo = *(const v8bf*)(rp + lhalf * 8);
      v8bf hi = *(const v8bf*)(rp + 16 + lhalf * 8);
#pragma unroll
      for (int i = 0; i < 8; i++) { a[mt][i] = lo[i]; a[mt][8 + i] = hi[i]; }
    }
#pragma unroll
    for (int nt = 0; nt < 4; nt++) {
      const bf16* cp = &Bs[cur][(wn + nt * 16 + l16) * PK + lhalf * 8];
      v16bf b;
      {
        v8bf b0 = *(const v8bf*)cp, b1 = *(const v8bf*)(cp + 16);
#pragma unroll
        for (int i = 0; i < 8; i++) { b[i] = b0[i]; b[8 + i] = b1[i]; }
      }
#pragma unroll
      for (int mt = 0; mt < 2; mt++) c0[mt][nt] = WMMA_BF16(a[mt], b, c0[mt][nt]);
      if (MODE == GEMM_GATEUP) {
        const bf16* cp2 = &B2s[cur * BN * PK + (wn + nt * 16 + l16) * PK + lhalf * 8];
        v16bf b2;
        v8bf b0 = *(const v8bf*)cp2, b1 = *(const v8bf*)(cp2 + 16);
#pragma unroll
        for (int i = 0; i < 8; i++) { b2[i] = b0[i]; b2[8 + i] = b1[i]; }
#pragma unroll
        for (int mt = 0; mt < 2; mt++) c1[mt][nt] = WMMA_BF16(a[mt], b2, c1[mt][nt]);
      }
    }
    __syncthreads();          // everyone done with buffer `cur` before it is re-filled
  }

  // ---- epilogue (C layout: lane holds n=l16, vgpr r holds m = r + lhalf*8)
#pragma unroll
  for (int mt = 0; mt < 2; mt++)
#pragma unroll
    for (int nt = 0; nt < 4; nt++) {
      const int col = n0 + wn + nt * 16 + l16;
#pragma unroll
      for (int r = 0; r < 8; r++) {
        const int row = m0 + wm + mt * 16 + r + lhalf * 8;
        const size_t idx = (size_t)row * N + col;
        if (MODE == GEMM_OUT_F32) {
          outF[idx] = c0[mt][nt][r];
        } else if (MODE == GEMM_ACC_F32) {
          outF[idx] += c0[mt][nt][r];
        } else {
          float g = c0[mt][nt][r], u = c1[mt][nt][r];
          float t = 0.7978845608028654f * (g + 0.044715f * g * g * g);
          float gel = 0.5f * g * (1.0f + tanhf(t));
          outB[idx] = (bf16)(gel * u);
        }
      }
    }
}

// ================================================================ RoPE (+1/sqrt(HD) on Q) + V-transpose
__global__ __launch_bounds__(256) void rope_kernel(const float* __restrict__ qf,
                                                   const float* __restrict__ kf,
                                                   const float* __restrict__ vf,
                                                   const float* __restrict__ cosp,
                                                   const float* __restrict__ sinp,
                                                   bf16* __restrict__ qb,
                                                   bf16* __restrict__ kb,
                                                   bf16* __restrict__ vt) {
  const int tkn = blockIdx.x;               // 0 .. B*S-1
  const int s = tkn % SS, b = tkn / SS;
  constexpr int HHF = HDIM / 2;             // 128
  constexpr float SCALE = 0.0625f;          // HD^-0.5
  const float* crow = cosp + (size_t)s * HHF;
  const float* srow = sinp + (size_t)s * HHF;
  for (int i = threadIdx.x; i < NHEAD * HHF; i += blockDim.x) {  // Q rope, pre-scaled
    const int hd = i / HHF, d = i % HHF;
    const size_t base = ((size_t)tkn * NHEAD + hd) * HDIM;
    const float x1 = qf[base + d], x2 = qf[base + d + HHF];
    const float c = crow[d], sn = srow[d];
    qb[base + d]       = (bf16)((x1 * c - x2 * sn) * SCALE);
    qb[base + d + HHF] = (bf16)((x1 * sn + x2 * c) * SCALE);
  }
  for (int d = threadIdx.x; d < HHF; d += blockDim.x) {          // K rope (NKV=1)
    const size_t base = (size_t)tkn * HDIM;
    const float x1 = kf[base + d], x2 = kf[base + d + HHF];
    const float c = crow[d], sn = srow[d];
    kb[base + d]       = (bf16)(x1 * c - x2 * sn);
    kb[base + d + HHF] = (bf16)(x1 * sn + x2 * c);
  }
  for (int d = threadIdx.x; d < HDIM; d += blockDim.x)           // V -> [b][hd][s]
    vt[((size_t)b * HDIM + d) * SS + s] = (bf16)vf[(size_t)tkn * HDIM + d];
}

// ================================================================ Flash attention (one wave / 16-row Q block)
// Q tile kept in LDS (async-staged) so the wave stays under 256 VGPRs.
__global__ __launch_bounds__(32) void attention_kernel(const bf16* __restrict__ Qp,
                                                       const bf16* __restrict__ Kp,
                                                       const bf16* __restrict__ Vtp,
                                                       bf16* __restrict__ Op) {
  constexpr int QP = 264;                   // padded Q row pitch (bank spread)
  const int b = blockIdx.z, h = blockIdx.y, qb = blockIdx.x;
  const int lane = threadIdx.x, lhalf = lane >> 4, l16 = lane & 15;
  const int q0 = qb * 16;
  __shared__ bf16 qlds[16 * QP];
  __shared__ bf16 plds[16 * 40];

  // ---- async-stage Q tile 16x256 (raw copy; already scaled by RoPE kernel)
  {
    const bf16* qbase = Qp + ((size_t)(b * SS + q0) * NHEAD + h) * HDIM;
#pragma unroll
    for (int j = 0; j < 8; j++) {
      const int chunk = j * 32 + lane;            // 256 chunks of 16 elems
      const int row = chunk >> 4, col = (chunk & 15) * 16;
      async_copy16(qbase + (size_t)row * (NHEAD * HDIM) + col, &qlds[row * QP + col]);
    }
    wait_async0();
  }

  v8f o[16];
#pragma unroll
  for (int t = 0; t < 16; t++) o[t] = v8f{};
  float mrun[8], lrun[8];
#pragma unroll
  for (int r = 0; r < 8; r++) { mrun[r] = -3.0e38f; lrun[r] = 0.f; }

  const bf16* Kb = Kp + (size_t)b * SS * HDIM;     // [s][HD]
  const bf16* Vb = Vtp + (size_t)b * HDIM * SS;    // [hd][s]
  const int nkb = (q0 + 15) / 32;                  // causal: blocks 0..nkb

  for (int kb = 0; kb <= nkb; kb++) {
    const int key00 = kb * 32;
    // ---- scores: two 16x16 tiles; Q A-frags re-read from LDS per chunk
    v8f s0 = v8f{}, s1 = v8f{};
    const bf16* krow0 = Kb + (size_t)(key00 + l16) * HDIM;
    const bf16* krow1 = krow0 + (size_t)16 * HDIM;
#pragma unroll
    for (int c = 0; c < 8; c++) {
      v16bf aqc;
      {
        const bf16* rp = &qlds[l16 * QP + c * 32];
        v8bf lo = *(const v8bf*)(rp + lhalf * 8);
        v8bf hi = *(const v8bf*)(rp + 16 + lhalf * 8);
#pragma unroll
        for (int i = 0; i < 8; i++) { aqc[i] = lo[i]; aqc[8 + i] = hi[i]; }
      }
      {
        const bf16* p = krow0 + c * 32 + lhalf * 8;
        v16bf bk; v8bf b0 = *(const v8bf*)p, b1 = *(const v8bf*)(p + 16);
#pragma unroll
        for (int i = 0; i < 8; i++) { bk[i] = b0[i]; bk[8 + i] = b1[i]; }
        s0 = WMMA_BF16(aqc, bk, s0);
      }
      {
        const bf16* p = krow1 + c * 32 + lhalf * 8;
        v16bf bk; v8bf b0 = *(const v8bf*)p, b1 = *(const v8bf*)(p + 16);
#pragma unroll
        for (int i = 0; i < 8; i++) { bk[i] = b0[i]; bk[8 + i] = b1[i]; }
        s1 = WMMA_BF16(aqc, bk, s1);
      }
    }
    // ---- causal mask + online softmax (rows live in half-waves; 16-lane reductions)
#pragma unroll
    for (int r = 0; r < 8; r++) {
      const int qrow = q0 + r + lhalf * 8;
      float v0 = s0[r], v1 = s1[r];
      if (key00 + l16 > qrow)      v0 = -1e9f;
      if (key00 + 16 + l16 > qrow) v1 = -1e9f;
      float tmax = fmaxf(v0, v1);
      for (int off = 8; off > 0; off >>= 1) tmax = fmaxf(tmax, __shfl_xor(tmax, off, 32));
      const float mnew = fmaxf(mrun[r], tmax);
      const float p0 = __expf(v0 - mnew), p1 = __expf(v1 - mnew);
      const float corr = __expf(mrun[r] - mnew);
      float rs = p0 + p1;
      for (int off = 8; off > 0; off >>= 1) rs += __shfl_xor(rs, off, 32);
      lrun[r] = lrun[r] * corr + rs;
      mrun[r] = mnew;
#pragma unroll
      for (int t = 0; t < 16; t++) o[t][r] *= corr;
      plds[(r + lhalf * 8) * 40 + l16]      = (bf16)p0;   // C-layout -> LDS
      plds[(r + lhalf * 8) * 40 + 16 + l16] = (bf16)p1;
    }
    asm volatile("s_wait_dscnt 0" ::: "memory");          // LDS turnaround
    // ---- P as A-fragment (16x32)
    v16bf ap;
    {
      const bf16* rp = &plds[l16 * 40];
      v8bf lo = *(const v8bf*)(rp + lhalf * 8);
      v8bf hi = *(const v8bf*)(rp + 16 + lhalf * 8);
#pragma unroll
      for (int i = 0; i < 8; i++) { ap[i] = lo[i]; ap[8 + i] = hi[i]; }
    }
    // ---- O += P x V  (V pre-transposed: contiguous along keys)
#pragma unroll
    for (int t = 0; t < 16; t++) {
      const bf16* vp = Vb + (size_t)(t * 16 + l16) * SS + key00 + lhalf * 8;
      v16bf bv;
      v8bf b0 = *(const v8bf*)vp, b1 = *(const v8bf*)(vp + 16);
#pragma unroll
      for (int i = 0; i < 8; i++) { bv[i] = b0[i]; bv[8 + i] = b1[i]; }
      o[t] = WMMA_BF16(ap, bv, o[t]);
    }
  }
  // ---- normalize + store bf16 [b,s,nh,hd]
#pragma unroll
  for (int t = 0; t < 16; t++)
#pragma unroll
    for (int r = 0; r < 8; r++) {
      const float v = o[t][r] / lrun[r];
      const size_t idx = ((size_t)(b * SS + q0 + r + lhalf * 8) * NHEAD + h) * HDIM + t * 16 + l16;
      Op[idx] = (bf16)v;
    }
}

// ================================================================ launch
extern "C" void kernel_launch(void* const* d_in, const int* in_sizes, int n_in,
                              void* d_out, int out_size, void* d_ws, size_t ws_size,
                              hipStream_t stream) {
  const float* hidden = (const float*)d_in[0];
  const float* cosp   = (const float*)d_in[1];
  const float* sinp   = (const float*)d_in[2];
  // d_in[3] mask, d_in[4] k_cache, d_in[5] v_cache: causal mask / identity kv indices folded in
  const float* wq  = (const float*)d_in[6];
  const float* wk  = (const float*)d_in[7];
  const float* wv  = (const float*)d_in[8];
  const float* wo  = (const float*)d_in[9];
  const float* wg  = (const float*)d_in[10];
  const float* wu  = (const float*)d_in[11];
  const float* wd  = (const float*)d_in[12];
  const float* ln1 = (const float*)d_in[13];
  const float* ln2 = (const float*)d_in[14];
  const float* nw  = (const float*)d_in[15];
  float* out = (float*)d_out;

  // ---- workspace carve-out
  char* p = (char*)d_ws;
  auto take = [&](size_t bytes) { char* q = p; p += (bytes + 255) & ~size_t(255); return q; };
  float* h_f32 = (float*)take((size_t)MTOK * HH * 4);
  float* q_f32 = (float*)take((size_t)MTOK * NHEAD * HDIM * 4);
  float* k_f32 = (float*)take((size_t)MTOK * HDIM * 4);
  float* v_f32 = (float*)take((size_t)MTOK * HDIM * 4);
  bf16*  xn    = (bf16*) take((size_t)MTOK * HH * 2);
  bf16*  qbf   = (bf16*) take((size_t)MTOK * NHEAD * HDIM * 2);
  bf16*  kbf   = (bf16*) take((size_t)MTOK * HDIM * 2);
  bf16*  vtb   = (bf16*) take((size_t)MTOK * HDIM * 2);
  bf16*  attnb = (bf16*) take((size_t)MTOK * NHEAD * HDIM * 2);
  bf16*  actb  = (bf16*) take((size_t)MTOK * IDIM * 2);
  // bf16 weight shadows (converted once per launch; halves GEMM weight traffic)
  const size_t NQ = (size_t)LAYERS * NHEAD * HDIM * HH;   // wq / wo elems
  const size_t NK = (size_t)LAYERS * HDIM * HH;           // wk / wv elems
  const size_t NG = (size_t)LAYERS * IDIM * HH;           // wg / wu / wd elems
  bf16* wqb = (bf16*)take(NQ * 2);
  bf16* wkb = (bf16*)take(NK * 2);
  bf16* wvb = (bf16*)take(NK * 2);
  bf16* wob = (bf16*)take(NQ * 2);
  bf16* wgb = (bf16*)take(NG * 2);
  bf16* wub = (bf16*)take(NG * 2);
  bf16* wdb = (bf16*)take(NG * 2);
  if ((size_t)(p - (char*)d_ws) > ws_size) return;  // insufficient scratch

  hipMemcpyAsync(h_f32, hidden, (size_t)MTOK * HH * 4, hipMemcpyDeviceToDevice, stream);

  const dim3 blk256(256), blk32(32);
  cvt_bf16_kernel<<<2048, blk256, 0, stream>>>((const float4*)wq, wqb, NQ / 4);
  cvt_bf16_kernel<<<2048, blk256, 0, stream>>>((const float4*)wk, wkb, NK / 4);
  cvt_bf16_kernel<<<2048, blk256, 0, stream>>>((const float4*)wv, wvb, NK / 4);
  cvt_bf16_kernel<<<2048, blk256, 0, stream>>>((const float4*)wo, wob, NQ / 4);
  cvt_bf16_kernel<<<4096, blk256, 0, stream>>>((const float4*)wg, wgb, NG / 4);
  cvt_bf16_kernel<<<4096, blk256, 0, stream>>>((const float4*)wu, wub, NG / 4);
  cvt_bf16_kernel<<<4096, blk256, 0, stream>>>((const float4*)wd, wdb, NG / 4);

  for (int l = 0; l < LAYERS; l++) {
    const size_t oqk = (size_t)l * NHEAD * HDIM * HH;
    const size_t okv = (size_t)l * HDIM * HH;
    rmsnorm_kernel<bf16><<<MTOK, blk256, 0, stream>>>(h_f32, ln1 + (size_t)l * HH, xn, HH);

    gemm_kernel<GEMM_OUT_F32><<<dim3((NHEAD * HDIM) / 128, MTOK / 128), blk256, 0, stream>>>(
        xn, wqb + oqk, nullptr, q_f32, nullptr, MTOK, NHEAD * HDIM, HH);
    gemm_kernel<GEMM_OUT_F32><<<dim3(HDIM / 128, MTOK / 128), blk256, 0, stream>>>(
        xn, wkb + okv, nullptr, k_f32, nullptr, MTOK, HDIM, HH);
    gemm_kernel<GEMM_OUT_F32><<<dim3(HDIM / 128, MTOK / 128), blk256, 0, stream>>>(
        xn, wvb + okv, nullptr, v_f32, nullptr, MTOK, HDIM, HH);

    rope_kernel<<<MTOK, blk256, 0, stream>>>(q_f32, k_f32, v_f32, cosp, sinp, qbf, kbf, vtb);

    attention_kernel<<<dim3(SS / 16, NHEAD, BB), blk32, 0, stream>>>(qbf, kbf, vtb, attnb);

    gemm_kernel<GEMM_ACC_F32><<<dim3(HH / 128, MTOK / 128), blk256, 0, stream>>>(
        attnb, wob + oqk, nullptr, h_f32, nullptr, MTOK, HH, NHEAD * HDIM);

    rmsnorm_kernel<bf16><<<MTOK, blk256, 0, stream>>>(h_f32, ln2 + (size_t)l * HH, xn, HH);

    gemm_kernel<GEMM_GATEUP><<<dim3(IDIM / 128, MTOK / 128), blk256, 0, stream>>>(
        xn, wgb + (size_t)l * IDIM * HH, wub + (size_t)l * IDIM * HH, nullptr, actb, MTOK, IDIM, HH);

    gemm_kernel<GEMM_ACC_F32><<<dim3(HH / 128, MTOK / 128), blk256, 0, stream>>>(
        actb, wdb + (size_t)l * IDIM * HH, nullptr, h_f32, nullptr, MTOK, HH, IDIM);
  }
  rmsnorm_kernel<float><<<MTOK, blk256, 0, stream>>>(h_f32, nw, out, HH);
}
